// APGIInspiredNetwork_49254684951090
// MI455X (gfx1250) — matrix operand order repaired and checked by reading, not compiled, not run on hardware
//
#include <hip/hip_runtime.h>
#include <math.h>

// ---------------------------------------------------------------------------
// APGI network forward, MI455X (gfx1250, wave32, WMMA bf16 16x16x32)
// One wave = 16 batch rows. Every dense layer -- including the small
// precision/threshold/GRU-input nets (zero-padded to 16x16x32 tiles) -- runs
// on v_wmma. Weights pre-converted/padded to bf16 into d_ws by prep kernels.
// All global staging is float4 (b128) since 16-row tiles are contiguous.
// ---------------------------------------------------------------------------

typedef __bf16 bf16;
typedef __attribute__((ext_vector_type(16))) __bf16 v16bf;
typedef __attribute__((ext_vector_type(8)))  float  v8f;
typedef __attribute__((ext_vector_type(4)))  float  v4f;

#define NP 44
struct Params { const float* p[NP]; };

enum {
  EW1, EB1, EW2, EB2, EW3, EB3,
  IW1, IB1, IW2, IB2, IW3, IB3,
  PEW1, PEB1, PEW2, PEB2,
  PIW1, PIB1, PIW2, PIB2,
  GWIH, GWHH, GBIH, GBHH,
  THW1, THB1, THW2, THB2,
  WSW, WSB,
  SOW1, SOB1, SOW2, SOB2,
  POW1, POB1, POW2, POB2,
  VAW1, VAB1, VAW2, VAB2,
  BETA, ALPHA
};

// bf16 weight arena offsets (elements) inside d_ws; all 32B-aligned.
constexpr int OFF_EW1  = 0;      // 128x128
constexpr int OFF_EW2  = 16384;  // 64x128
constexpr int OFF_EW3  = 24576;  // 32x64
constexpr int OFF_IW1  = 26624;  // 64x32
constexpr int OFF_IW2  = 28672;  // 32x64
constexpr int OFF_IW3  = 30720;  // 16x32
constexpr int OFF_WSW  = 31232;  // 64x64  (src 64x48, K-pad)
constexpr int OFF_SOW1 = 35328;  // 64x64
constexpr int OFF_SOW2 = 39424;  // 64x64
constexpr int OFF_POW1 = 43520;  // 32x64
constexpr int OFF_POW2 = 45568;  // 16x32
constexpr int OFF_VAW1 = 46080;  // 32x64
constexpr int OFF_PEW1 = 48128;  // 16x32
constexpr int OFF_PIW1 = 48640;  // 16x32  (src 8x16, N/K-pad)
constexpr int OFF_THW1 = 49152;  // 16x32  (src 16x8, K-pad)
constexpr int OFF_GWIH = 49664;  // 48x32  (src 48x2, K-pad)
constexpr int OFF_VAW2 = 51200;  // 16x32  (src 1x32, N-pad)
constexpr int WB_TOTAL = 51712;  // bf16 elements (103,424 bytes)

__device__ __forceinline__ float sigmoidf_(float x) { return 1.0f / (1.0f + __expf(-x)); }
__device__ __forceinline__ float softplusf_(float x) {
  return fmaxf(x, 0.0f) + log1pf(__expf(-fabsf(x)));   // jax softplus
}

// Weight prep: dst[Npad][Kpad] = (bf16)src[N][K], zero-padded.
__global__ void cvt_w_kernel(const float* __restrict__ src, bf16* __restrict__ dst,
                             int N, int K, int Npad, int Kpad) {
  int idx = blockIdx.x * blockDim.x + threadIdx.x;
  if (idx >= Npad * Kpad) return;
  int n = idx / Kpad, k = idx - n * Kpad;
  dst[idx] = (n < N && k < K) ? (bf16)src[n * K + k] : (bf16)0.0f;
}

// A fragment: 16x32 bf16 from an LDS fp32 tile (row-major, leading dim ldk).
// Lane L -> row = L&15, kg = L>>4; element s: K = kg*16+s (contiguous)
__device__ __forceinline__ v16bf load_A16(const float* act, int ldk, int k0, int lane) {
  int row = lane & 15, kg = lane >> 4;
  const float* p = act + row * ldk + k0 + kg * 16;
  v16bf a;
#pragma unroll
  for (int j = 0; j < 8; ++j) a[j] = (bf16)p[j];
#pragma unroll
  for (int j = 0; j < 8; ++j) a[8 + j] = (bf16)p[8 + j];
  return a;
}

// out[16][N] (LDS fp32, ld ldo) = act[16][K] @ Wbf[N][K]^T + bias, opt ReLU.
// A fragments hoisted; B fragment = one aligned 32B bf16 vector load.
// NB: number of valid bias entries (zero bias for padded output cols).
template<int N, int K, bool RELU, int NB = N>
__device__ __forceinline__ void gemm16(const float* __restrict__ actIn, int ldk,
                                       const bf16* __restrict__ W,
                                       const float* __restrict__ bias,
                                       float* __restrict__ outT, int ldo, int lane) {
  constexpr int KT = K / 32;
  v16bf a[KT];
#pragma unroll
  for (int kt = 0; kt < KT; ++kt) a[kt] = load_A16(actIn, ldk, kt * 32, lane);

  const int col = lane & 15, kg = lane >> 4;
#pragma unroll
  for (int n0 = 0; n0 < N; n0 += 16) {
    v8f acc = {0.f, 0.f, 0.f, 0.f, 0.f, 0.f, 0.f, 0.f};
#pragma unroll
    for (int kt = 0; kt < KT; ++kt) {
      v16bf b = *(const v16bf*)(W + (size_t)(n0 + col) * K + kt * 32 + kg * 16);
      acc = __builtin_amdgcn_wmma_f32_16x16x32_bf16(false, a[kt], false, b,
                                                    (short)0, acc, false, false);
    }
    float bv;
    if constexpr (NB >= N) bv = bias[n0 + col];
    else bv = (n0 + col < NB) ? bias[n0 + col] : 0.0f;
#pragma unroll
    for (int j = 0; j < 8; ++j) {
      float v = acc[j] + bv;
      if (RELU) v = fmaxf(v, 0.0f);
      outT[(size_t)(j + kg * 8) * ldo + n0 + col] = v;
    }
  }
}

#define WPB 2  // waves per block

__global__ __launch_bounds__(WPB * 32)
void apgi_fwd_kernel(const float* __restrict__ extero,
                     const float* __restrict__ intero,
                     const float* __restrict__ ctx,
                     Params P, const bf16* __restrict__ wb,
                     float* __restrict__ out, int nB) {
  __shared__ __align__(16) float lds[WPB][2][2048];   // per wave: 2x 16x128 fp32
  __shared__ float s_ign[WPB][16];

  const int wave = threadIdx.x >> 5;
  const int lane = threadIdx.x & 31;
  const int r0 = (blockIdx.x * WPB + wave) * 16;
  if (r0 >= nB) return;

  float* A    = lds[wave][0];
  float* Bb   = lds[wave][1];
  v4f*   A4   = (v4f*)A;
  v4f*   Bb4  = (v4f*)Bb;
  float* ign  = s_ign[wave];
  const v4f z4 = {0.f, 0.f, 0.f, 0.f};

  const size_t nb = (size_t)nB;
  float* out_policy = out;             // [B,16]
  float* out_value  = out + nb * 16;
  float* out_ign    = out + nb * 17;
  float* out_S      = out + nb * 18;
  float* out_theta  = out + nb * 19;
  float* out_PiE    = out + nb * 20;
  float* out_PiI    = out + nb * 21;
  float* out_som    = out + nb * 22;   // [B,64]
  float* out_gated  = out + nb * 86;   // [B,64]

  // ---- extero encoder: 128 -> 128 -> 64 -> 32 ------------------------------
  {
    const v4f* x4 = (const v4f*)(extero + (size_t)r0 * 128);  // 16 rows contiguous
    for (int t = lane; t < 512; t += 32) A4[t] = x4[t];
  }
  gemm16<128, 128, true >(A, 128, wb + OFF_EW1, P.p[EB1], Bb, 128, lane);
  gemm16< 64, 128, true >(Bb, 128, wb + OFF_EW2, P.p[EB2], A,  64, lane);
  gemm16< 32,  64, false>(A,  64, wb + OFF_EW3, P.p[EB3], Bb, 32, lane);  // encE

  // ---- intero encoder: 32 -> 64 -> 32 -> 16 --------------------------------
  {
    const v4f* x4 = (const v4f*)(intero + (size_t)r0 * 32);
    for (int t = lane; t < 128; t += 32) A4[t] = x4[t];
  }
  gemm16<64, 32, true >(A,        32, wb + OFF_IW1, P.p[IB1], A + 512,  64, lane);
  gemm16<32, 64, true >(A + 512,  64, wb + OFF_IW2, P.p[IB2], A + 1536, 32, lane);
  gemm16<16, 32, false>(A + 1536, 32, wb + OFF_IW3, P.p[IB3], Bb + 512, 16, lane); // encI

  const float* encE  = Bb;                    // [16][32]
  const float* encI  = Bb + 512;              // [16][16]
  const v4f*   encE4 = (const v4f*)encE;
  const v4f*   encI4 = (const v4f*)encI;

  // ---- precision / threshold / GRU-input nets via padded WMMA tiles --------
  // peH = relu(encE @ pe_w1^T + pe_b1) -> A[512..768), ld 16
  gemm16<16, 32, true>(encE, 32, wb + OFF_PEW1, P.p[PEB1], A + 512, 16, lane);

  // pi A-tile: [16][32] = [encI | 0] -> A[0..512)
  for (int t = lane; t < 128; t += 32) {
    int rr = t >> 3, q = t & 7;
    A4[t] = (q < 4) ? encI4[rr * 4 + q] : z4;
  }
  // piH (8 valid rows padded to 16) -> A[768..1024), ld 16
  gemm16<16, 32, true, 8>(A, 32, wb + OFF_PIW1, P.p[PIB1], A + 768, 16, lane);

  // ctx A-tile: [16][32] = [ctx(8) | 0] -> A[0..512)
  {
    const v4f* c4 = (const v4f*)(ctx + (size_t)r0 * 8);   // 16 rows contiguous
    for (int t = lane; t < 128; t += 32) {
      int rr = t >> 3, q = t & 7;
      A4[t] = (q < 2) ? c4[rr * 2 + q] : z4;
    }
  }
  // thH -> A[1024..1280), ld 16
  gemm16<16, 32, true>(A, 32, wb + OFF_THW1, P.p[THB1], A + 1024, 16, lane);

  // ---- per-row: norms, Pi_e/Pi_i, theta, surprise inputs -------------------
  float PiE = 0.f, PiI = 0.f, theta = 0.f;
  if (lane < 16) {
    const int r = lane;
    const float* xe = encE + r * 32;
    const float* xi = encI + r * 16;
    float ee = 0.f, ei = 0.f;
#pragma unroll
    for (int k = 0; k < 32; ++k) ee += xe[k] * xe[k];
#pragma unroll
    for (int k = 0; k < 16; ++k) ei += xi[k] * xi[k];

    float pe = P.p[PEB2][0];
#pragma unroll
    for (int h = 0; h < 16; ++h) pe += P.p[PEW2][h] * A[512 + r * 16 + h];
    PiE = softplusf_(pe);

    float pi = P.p[PIB2][0];
#pragma unroll
    for (int h = 0; h < 8; ++h) pi += P.p[PIW2][h] * A[768 + r * 16 + h];
    PiI = softplusf_(pi);

    float th = P.p[THB2][0];
#pragma unroll
    for (int h = 0; h < 16; ++h) th += P.p[THW2][h] * A[1024 + r * 16 + h];
    theta = sigmoidf_(th);

    const float s0 = PiE * sqrtf(ee);
    const float s1 = fabsf(P.p[BETA][0]) * PiI * sqrtf(ei);
    // surprise A-tile row: [s0, s1, 0..0]; cols 8..31 already zero (ctx pad)
    v4f sv = {s0, s1, 0.f, 0.f};
    A4[r * 8 + 0] = sv;
    A4[r * 8 + 1] = z4;
  }

  // gi = surprise @ gru_wih^T + b_ih  -> A[1280..2048), ld 48 (3 WMMA tiles)
  gemm16<48, 32, false>(A, 32, wb + OFF_GWIH, P.p[GBIH], A + 1280, 48, lane);

  // ---- per-row GRU nonlinearity (h0 = 0 => gh = b_hh), ignition ------------
  if (lane < 16) {
    const int r = lane;
    const float* gi = A + 1280 + r * 48;
    const float* bhh = P.p[GBHH];
    float S2 = 0.f;
#pragma unroll
    for (int j = 0; j < 16; ++j) {
      float rg = sigmoidf_(gi[j] + bhh[j]);
      float zg = sigmoidf_(gi[16 + j] + bhh[16 + j]);
      float ng = tanhf(gi[32 + j] + rg * bhh[32 + j]);
      float hj = (1.0f - zg) * ng;
      S2 += hj * hj;
    }
    const float S = sqrtf(S2);
    const float ig = sigmoidf_(fabsf(P.p[ALPHA][0]) * (S - theta));
    ign[r] = ig;

    const size_t gr = (size_t)(r0 + r);
    out_ign[gr]   = ig;
    out_S[gr]     = S;
    out_theta[gr] = theta;
    out_PiE[gr]   = PiE;
    out_PiI[gr]   = PiI;
  }

  // ---- combined [16][64] = [encE(32) | encI(16) | 0(16)] -> A[0..1024) -----
  for (int t = lane; t < 256; t += 32) {
    int rr = t >> 4, q = t & 15;
    A4[t] = (q < 8) ? encE4[rr * 8 + q]
          : (q < 12) ? encI4[rr * 4 + (q - 8)] : z4;
  }

  // workspace: 48 -> 64 (weights K-padded to 64)
  gemm16<64, 64, false>(A, 64, wb + OFF_WSW, P.p[WSB], Bb + 768, 64, lane);

  // gated = ignition * workspace -> A[0..1024) and global (contiguous tile)
  {
    v4f* og4 = (v4f*)(out_gated + (size_t)r0 * 64);
    for (int t = lane; t < 256; t += 32) {
      int rr = t >> 4;
      v4f g = Bb4[192 + t] * ign[rr];
      A4[t] = g;
      og4[t] = g;
    }
  }

  // somatic: 64 -> 64 (relu) -> 64
  gemm16<64, 64, true >(A,  64, wb + OFF_SOW1, P.p[SOB1], Bb,       64, lane);
  gemm16<64, 64, false>(Bb, 64, wb + OFF_SOW2, P.p[SOB2], A + 1024, 64, lane);

  // modulation; policy_in = gated * (1 + 0.3*sigmoid(somatic)) -> Bb[0..1024)
  {
    v4f* os4 = (v4f*)(out_som + (size_t)r0 * 64);
    for (int t = lane; t < 256; t += 32) {
      v4f som = A4[256 + t];
      os4[t] = som;
      v4f m;
#pragma unroll
      for (int i = 0; i < 4; ++i) m[i] = 1.0f + 0.3f * sigmoidf_(som[i]);
      Bb4[t] = A4[t] * m;
    }
  }

  // policy head: 64 -> 32 (relu) -> 16, softmax
  gemm16<32, 64, true >(Bb,       64, wb + OFF_POW1, P.p[POB1], A + 1024,  32, lane);
  gemm16<16, 32, false>(A + 1024, 32, wb + OFF_POW2, P.p[POB2], Bb + 1024, 16, lane);

  if (lane < 16) {
    const float* lg = Bb + 1024 + lane * 16;
    float m = lg[0];
#pragma unroll
    for (int j = 1; j < 16; ++j) m = fmaxf(m, lg[j]);
    float e[16], s = 0.f;
#pragma unroll
    for (int j = 0; j < 16; ++j) { e[j] = __expf(lg[j] - m); s += e[j]; }
    float inv = 1.0f / s;
    v4f* po4 = (v4f*)(out_policy + (size_t)(r0 + lane) * 16);
#pragma unroll
    for (int q = 0; q < 4; ++q) {
      v4f pv = {e[4 * q] * inv, e[4 * q + 1] * inv, e[4 * q + 2] * inv, e[4 * q + 3] * inv};
      po4[q] = pv;
    }
  }

  // value head: 64 -> 32 (relu) -> 1 (va2 N-padded to 16, only col 0 valid)
  gemm16<32, 64, true>(A, 64, wb + OFF_VAW1, P.p[VAB1], Bb, 32, lane);
  gemm16<16, 32, false, 1>(Bb, 32, wb + OFF_VAW2, P.p[VAB2], Bb + 512, 16, lane);
  if (lane < 16)
    out_value[(size_t)(r0 + lane)] = Bb[512 + lane * 16];
}

extern "C" void kernel_launch(void* const* d_in, const int* in_sizes, int n_in,
                              void* d_out, int out_size, void* d_ws, size_t ws_size,
                              hipStream_t stream) {
  const float* extero = (const float*)d_in[0];
  const float* intero = (const float*)d_in[1];
  const float* ctx    = (const float*)d_in[2];

  Params P;
  for (int i = 0; i < NP; ++i) P.p[i] = (const float*)d_in[3 + i];

  bf16* wb = (bf16*)d_ws;

  // Per-launch bf16 weight prep (same stream -> ordered before main kernel).
  struct WCfg { int pidx, off, N, K, Npad, Kpad; };
  const WCfg cfg[17] = {
    {EW1,  OFF_EW1,  128, 128, 128, 128},
    {EW2,  OFF_EW2,   64, 128,  64, 128},
    {EW3,  OFF_EW3,   32,  64,  32,  64},
    {IW1,  OFF_IW1,   64,  32,  64,  32},
    {IW2,  OFF_IW2,   32,  64,  32,  64},
    {IW3,  OFF_IW3,   16,  32,  16,  32},
    {WSW,  OFF_WSW,   64,  48,  64,  64},
    {SOW1, OFF_SOW1,  64,  64,  64,  64},
    {SOW2, OFF_SOW2,  64,  64,  64,  64},
    {POW1, OFF_POW1,  32,  64,  32,  64},
    {POW2, OFF_POW2,  16,  32,  16,  32},
    {VAW1, OFF_VAW1,  32,  64,  32,  64},
    {PEW1, OFF_PEW1,  16,  32,  16,  32},
    {PIW1, OFF_PIW1,   8,  16,  16,  32},
    {THW1, OFF_THW1,  16,   8,  16,  32},
    {GWIH, OFF_GWIH,  48,   2,  48,  32},
    {VAW2, OFF_VAW2,   1,  32,  16,  32},
  };
  for (int i = 0; i < 17; ++i) {
    int total = cfg[i].Npad * cfg[i].Kpad;
    int grid = (total + 255) / 256;
    cvt_w_kernel<<<grid, 256, 0, stream>>>(P.p[cfg[i].pidx], wb + cfg[i].off,
                                           cfg[i].N, cfg[i].K, cfg[i].Npad, cfg[i].Kpad);
  }

  const int nB = in_sizes[0] / 128;          // batch
  const int rowsPerBlock = WPB * 16;         // 32
  const int grid = (nB + rowsPerBlock - 1) / rowsPerBlock;

  apgi_fwd_kernel<<<grid, WPB * 32, 0, stream>>>(extero, intero, ctx, P, wb,
                                                 (float*)d_out, nB);
}
